// VectorQuantizerHiRes_65970697667496
// MI455X (gfx1250) — matrix operand order repaired and checked by reading, not compiled, not run on hardware
//
#include <hip/hip_runtime.h>
#include <hip/hip_bf16.h>

typedef _Float16 f16;
typedef __attribute__((ext_vector_type(8)))  _Float16 v8h;
typedef __attribute__((ext_vector_type(16))) _Float16 v16h;
typedef __attribute__((ext_vector_type(8)))  float    v8f;
typedef __attribute__((ext_vector_type(4)))  unsigned int v4u;
typedef __attribute__((ext_vector_type(4)))  unsigned int u32x4;
typedef __attribute__((ext_vector_type(8)))  int          i32x8;
typedef __attribute__((ext_vector_type(4)))  int          i32x4;

// Problem constants (from reference): z [16,256,64,64] f32, codebook [512,256] f32
#define BATCH 16
#define CDIM  256
#define HDIM  64
#define WDIM  64
#define KCODES 512
#define ZQ_ELEMS (BATCH * CDIM * HDIM * WDIM)  // 16777216
#define NUM_WG (BATCH * HDIM)                  // 1024 workgroups, 64 positions each
#define ZROW_STRIDE 264                        // 256 + 8 halves pad (528B, 16B multiple)

// ---------------------------------------------------------------------------
// Kernel 1: codebook f32 -> f16 (workspace) + per-code squared norms
// ---------------------------------------------------------------------------
__global__ __launch_bounds__(256) void vq_prep(const float* __restrict__ cb,
                                               f16* __restrict__ cbh,
                                               float* __restrict__ enorm) {
    int k = blockIdx.x;          // code index 0..511
    int c = threadIdx.x;         // channel 0..255
    float v = cb[k * CDIM + c];
    cbh[k * CDIM + c] = (f16)v;
    float s = v * v;
    #pragma unroll
    for (int mask = 1; mask < 32; mask <<= 1) s += __shfl_xor(s, mask, 32);
    __shared__ float rr[8];
    int lane = c & 31, wv = c >> 5;
    if (lane == 0) rr[wv] = s;
    __syncthreads();
    if (c == 0) {
        float t = 0.f;
        #pragma unroll
        for (int i = 0; i < 8; ++i) t += rr[i];
        enorm[k] = t;
    }
}

// ---------------------------------------------------------------------------
// Kernel 2: main VQ kernel. One WG per (b,h): 64 positions, 128 threads (4 waves).
// Full f16 codebook DMA'd by the Tensor Data Mover into LDS (256KB of the
// 320KB CDNA5 WGP LDS), overlapped with the z-tile transpose done by all waves.
// Each wave: 16 rows x 512 codes via v_wmma_f32_16x16x32_f16.
// ---------------------------------------------------------------------------
__global__ __launch_bounds__(128) void vq_main(const float* __restrict__ z,
                                               const float* __restrict__ cb,
                                               const f16*   __restrict__ cbh,
                                               const float* __restrict__ enorm,
                                               float* __restrict__ zq_out,
                                               float* __restrict__ idx_out,
                                               float* __restrict__ partials) {
    __shared__ alignas(16) f16 cb_lds[KCODES * CDIM];       // 262144 B
    __shared__ alignas(16) f16 z_lds[WDIM * ZROW_STRIDE];   //  33792 B
    __shared__ float enorm_lds[KCODES];                     //   2048 B
    __shared__ int   idx_lds[WDIM];                         //    256 B
    __shared__ float red[4];

    const int tid  = threadIdx.x;
    const int lane = tid & 31;
    const int wave = tid >> 5;
    const int b = blockIdx.x >> 6;
    const int h = blockIdx.x & 63;
    const long zbase = (long)b * (CDIM * HDIM * WDIM) + (long)h * WDIM;

    // ---- stage f16 codebook into LDS via Tensor Data Mover ----
#if __has_builtin(__builtin_amdgcn_tensor_load_to_lds)
    if (tid < 32) {
        // Tensor DMA descriptor (ISA 08_async_tensor §8), data_size=3 (8B units):
        // 2-D tensor: 512 rows x 64 units (512B) contiguous -> LDS, one shot.
        const unsigned long long ga = (unsigned long long)(uintptr_t)cbh;
        const unsigned lds_addr = (unsigned)(uintptr_t)&cb_lds[0]; // LDS byte offset
        u32x4 g0;
        g0[0] = 1u;                                  // count=1 (valid descriptor)
        g0[1] = lds_addr;                            // lds_addr
        g0[2] = (unsigned)(ga & 0xffffffffu);        // global_addr[31:0]
        g0[3] = (unsigned)((ga >> 32) & 0x01ffffffu) // global_addr[56:32]
                | 0x80000000u;                       // type=2 ("image") in [127:126]
        i32x8 g1;
        g1[0] = (3 << 16);          // workgroup_mask=0, data_size=3 (8B)
        g1[1] = (64 << 16);         // tensor_dim0[15:0]=64 units  (bits 63:48)
        g1[2] = (512 << 16);        // tensor_dim0 hi=0 | tensor_dim1 lo=512
        g1[3] = (64 << 16);         // tensor_dim1 hi=0 | tile_dim0=64
        g1[4] = 512;                // tile_dim1=512 rows, tile_dim2=0
        g1[5] = 64;                 // tensor_dim0_stride lo = 64 units (512B)
        g1[6] = 0;                  // stride0 hi | stride1 lo
        g1[7] = 0;                  // stride1 hi
        i32x4 g2 = {0, 0, 0, 0};    // dims 2+ unused (tile_dim3=0)
        i32x4 g3 = {0, 0, 0, 0};
        i32x8 g4 = {0, 0, 0, 0, 0, 0, 0, 0}; // extra group (clang-23 6-arg form)
        __builtin_amdgcn_tensor_load_to_lds(g0, g1, g2, g3, g4, 0);
        __builtin_amdgcn_s_wait_tensorcnt(0);
    }
#else
    {
        const v4u* src = (const v4u*)cbh;
        v4u* dst = (v4u*)cb_lds;
        for (int i = tid; i < (KCODES * CDIM) / 8; i += 128) dst[i] = src[i];
    }
#endif
    for (int i = tid; i < KCODES; i += 128) enorm_lds[i] = enorm[i];

    // ---- stage z tile (transpose NCHW -> [pos][c], f32 -> f16); overlaps TDM ----
    for (int i = tid; i < CDIM * WDIM; i += 128) {
        int c = i >> 6, w = i & 63;
        z_lds[w * ZROW_STRIDE + c] = (f16)z[zbase + (long)c * (HDIM * WDIM) + w];
    }
    __syncthreads();

    // ---- per-wave WMMA distance + argmin ----
    const int m = lane & 15;                 // A row / C column within tile
    const int hi = (lane & 16) ? 1 : 0;      // lane half
    // A operand: row = position wave*16 + m, 256 channels -> 8 chunks of K=32
    v16h a[8];
    {
        const int arow = (wave * 16 + m) * ZROW_STRIDE + hi * 8; // halves
        #pragma unroll
        for (int kc = 0; kc < 8; ++kc) {
            const v8h l0 = *(const v8h*)&z_lds[arow + kc * 32];
            const v8h l1 = *(const v8h*)&z_lds[arow + kc * 32 + 16];
            a[kc] = __builtin_shufflevector(l0, l1, 0,1,2,3,4,5,6,7,8,9,10,11,12,13,14,15);
        }
    }

    float best_d[8]; int best_i[8];
    #pragma unroll
    for (int r = 0; r < 8; ++r) { best_d[r] = 3.0e38f; best_i[r] = 0; }

    for (int nt = 0; nt < KCODES / 16; ++nt) {
        const int n = nt * 16 + m;           // code column this lane owns
        v8f acc = {};
        const int brow = n * CDIM + hi * 8;  // halves
        #pragma unroll
        for (int kc = 0; kc < 8; ++kc) {
            const v8h l0 = *(const v8h*)&cb_lds[brow + kc * 32];
            const v8h l1 = *(const v8h*)&cb_lds[brow + kc * 32 + 16];
            const v16h bm = __builtin_shufflevector(l0, l1, 0,1,2,3,4,5,6,7,8,9,10,11,12,13,14,15);
            acc = __builtin_amdgcn_wmma_f32_16x16x32_f16(
                false, a[kc], false, bm, (short)0, acc, false, false);
        }
        const float en = enorm_lds[n];
        #pragma unroll
        for (int r = 0; r < 8; ++r) {
            float d = en - 2.0f * acc[r];    // |z|^2 constant per row: dropped
            if (d < best_d[r]) { best_d[r] = d; best_i[r] = n; }
        }
    }

    // reduce across the 16 lanes of each half (column group), lowest-index ties
    #pragma unroll
    for (int r = 0; r < 8; ++r) {
        float bd = best_d[r]; int bi = best_i[r];
        #pragma unroll
        for (int mask = 1; mask < 16; mask <<= 1) {
            float od = __shfl_xor(bd, mask, 32);
            int   oi = __shfl_xor(bi, mask, 32);
            if (od < bd || (od == bd && oi < bi)) { bd = od; bi = oi; }
        }
        if (m == r) {
            const int pos = wave * 16 + r + hi * 8;  // C layout: rows r / r+8
            idx_lds[pos] = bi;
            idx_out[(long)blockIdx.x * WDIM + pos] = (float)bi;
        }
    }
    __syncthreads();

    // ---- gather dequant (f32 codebook), scatter to NCHW, commitment partial ----
    float lsum = 0.f;
    for (int i = tid; i < CDIM * WDIM; i += 128) {
        int c = i >> 6, w = i & 63;
        const int code = idx_lds[w];
        const float q = cb[code * CDIM + c];
        const long gi = zbase + (long)c * (HDIM * WDIM) + w;
        zq_out[gi] = q;                        // straight-through == z_q numerically
        const float diff = q - z[gi];
        lsum += diff * diff;
    }
    #pragma unroll
    for (int mask = 1; mask < 32; mask <<= 1) lsum += __shfl_xor(lsum, mask, 32);
    if (lane == 0) red[wave] = lsum;
    __syncthreads();
    if (tid == 0) partials[blockIdx.x] = red[0] + red[1] + red[2] + red[3];
}

// ---------------------------------------------------------------------------
// Kernel 3: deterministic loss finalize: 0.25 * sum(partials) / numel
// ---------------------------------------------------------------------------
__global__ __launch_bounds__(256) void vq_loss(const float* __restrict__ partials,
                                               float* __restrict__ loss_out) {
    const int tid = threadIdx.x;
    float s = partials[tid] + partials[tid + 256] + partials[tid + 512] + partials[tid + 768];
    #pragma unroll
    for (int mask = 1; mask < 32; mask <<= 1) s += __shfl_xor(s, mask, 32);
    __shared__ float rr[8];
    int lane = tid & 31, wv = tid >> 5;
    if (lane == 0) rr[wv] = s;
    __syncthreads();
    if (tid == 0) {
        float t = 0.f;
        #pragma unroll
        for (int i = 0; i < 8; ++i) t += rr[i];
        loss_out[0] = 0.25f * t / (float)ZQ_ELEMS;
    }
}

// ---------------------------------------------------------------------------
extern "C" void kernel_launch(void* const* d_in, const int* in_sizes, int n_in,
                              void* d_out, int out_size, void* d_ws, size_t ws_size,
                              hipStream_t stream) {
    (void)in_sizes; (void)n_in; (void)out_size; (void)ws_size;
    const float* z  = (const float*)d_in[0];   // [16,256,64,64]
    const float* cb = (const float*)d_in[1];   // [512,256]

    // workspace layout: enorm f32[512] | cbh f16[512*256] | partials f32[1024]
    float* ws_enorm = (float*)d_ws;
    f16*   ws_cbh   = (f16*)((char*)d_ws + 2048);
    float* ws_part  = (float*)((char*)d_ws + 2048 + KCODES * CDIM * (int)sizeof(f16));

    // output layout: z_q [16777216] | loss [1] | indices-as-float [65536]
    float* zq   = (float*)d_out;
    float* loss = zq + ZQ_ELEMS;
    float* idxf = zq + ZQ_ELEMS + 1;

    vq_prep<<<KCODES, 256, 0, stream>>>(cb, ws_cbh, ws_enorm);
    vq_main<<<NUM_WG, 128, 0, stream>>>(z, cb, ws_cbh, ws_enorm, zq, idxf, ws_part);
    vq_loss<<<1, 256, 0, stream>>>(ws_part, loss);
}
